// GlobalRouters_25864293056819
// MI455X (gfx1250) — matrix operand order repaired and checked by reading, not compiled, not run on hardware
//
#include <hip/hip_runtime.h>
#include <hip/hip_bf16.h>
#include <math.h>

typedef __attribute__((ext_vector_type(16))) __bf16 v16bf;
typedef __attribute__((ext_vector_type(8)))  __bf16 v8bf;
typedef __attribute__((ext_vector_type(8)))  float  v8f;

#define SEQ   4096
#define DIM   2048
#define DSZ   64
#define NTOT  4096
#define BATCH 4

__device__ __forceinline__ __bf16 f2bf(float f) { return (__bf16)f; }

__device__ __forceinline__ v16bf cat8(v8bf a, v8bf b) {
  return __builtin_shufflevector(a, b, 0,1,2,3,4,5,6,7,8,9,10,11,12,13,14,15);
}

// ---------------------------------------------------------------------------
// Prep: zero accumulators, transpose W -> Wt[n][k] (bf16, K-contiguous rows),
// row-normalize neuron_emb -> embn[n][k] (bf16). Both land in the exact WMMA
// B-operand friendly layout (N-major, K-contiguous 32B chunks).
// ---------------------------------------------------------------------------
__global__ void prep_kernel(const float* __restrict__ W,
                            const float* __restrict__ emb,
                            __bf16* __restrict__ Wt,
                            __bf16* __restrict__ embn,
                            float* __restrict__ accum) {
  const int t = blockIdx.x * blockDim.x + threadIdx.x;
  const int nth = gridDim.x * blockDim.x;              // 65536

  for (int i = t; i < BATCH * NTOT; i += nth) accum[i] = 0.f;

  for (int i = t; i < DSZ * DIM; i += nth) {
    const int n = i / DIM, k = i % DIM;
    Wt[i] = f2bf(W[k * DSZ + n]);
  }

  // 16 lanes cooperate per embedding row
  const int row = t >> 4;
  const int l16 = t & 15;
  const int rstep = nth >> 4;
  for (int n = row; n < NTOT; n += rstep) {
    const float* er = emb + (size_t)n * DSZ;
    float s = 0.f;
    for (int k = l16; k < DSZ; k += 16) { float v = er[k]; s += v * v; }
    for (int m = 8; m >= 1; m >>= 1) s += __shfl_xor(s, m, 16);
    const float inv = rsqrtf(s);
    for (int k = l16; k < DSZ; k += 16) embn[(size_t)n * DSZ + k] = f2bf(er[k] * inv);
  }
}

// ---------------------------------------------------------------------------
// Fused main kernel: one block = one (batch, 16-row sequence tile).
// 128 threads = 4 wave32s; wave w owns h N-tile w in stage 1 and every 4th
// neuron tile in stage 2.
// ---------------------------------------------------------------------------
__global__ void __launch_bounds__(128)
main_kernel(const float* __restrict__ x,
            const float* __restrict__ importance,
            const __bf16* __restrict__ Wt,
            const float* __restrict__ bias,
            const __bf16* __restrict__ embn,
            float* __restrict__ accum) {
  __shared__ __bf16 xbf[16 * 520];   // 512-wide K-chunk of x tile, bf16, padded
  __shared__ __bf16 hbf[16 * 72];    // h tile (16 x 64), padded stride
  __shared__ float  rowSum[48];      // per-segment per-row softmax denominators
  __shared__ float  wrow[48];        // importance / denom

  const int tid  = threadIdx.x;
  const int w    = tid >> 5;         // wave id 0..3
  const int L    = tid & 31;         // lane
  const int mrow = L & 15;
  const int half = L >> 4;

  const int tileIdx = blockIdx.x & 255;
  const int b       = blockIdx.x >> 8;
  const int s0      = tileIdx * 16;

  if (tid < 48) rowSum[tid] = 0.f;

  // ------------------- stage 1: h = x @ W + b via WMMA -------------------
  const float* xbase = x + ((size_t)b * SEQ + s0) * DIM;
  const __bf16* wtcol = Wt + (size_t)(w * 16 + mrow) * DIM + half * 16;

  v8f hacc = {};
  for (int kc = 0; kc < 4; ++kc) {            // 4 chunks of 512 K each
    // cooperative stage of x[16][512] -> bf16 LDS (coalesced float4 loads)
    for (int i = tid; i < 2048; i += 128) {
      const int idx = i * 4;
      const int r = idx >> 9;                 // /512
      const int c = idx & 511;
      const float4 v = *(const float4*)(xbase + (size_t)r * DIM + kc * 512 + c);
      __bf16* dst = &xbf[r * 520 + c];
      dst[0] = f2bf(v.x); dst[1] = f2bf(v.y); dst[2] = f2bf(v.z); dst[3] = f2bf(v.w);
    }
    __syncthreads();
    #pragma unroll 4
    for (int ks = 0; ks < 16; ++ks) {
      const int kb = ks * 32;
      v8bf a0 = *(const v8bf*)&xbf[mrow * 520 + kb + half * 16];
      v8bf a1 = *(const v8bf*)&xbf[mrow * 520 + kb + half * 16 + 8];
      v16bf a  = cat8(a0, a1);
      v16bf bm = *(const v16bf*)(wtcol + kc * 512 + kb);
      hacc = __builtin_amdgcn_wmma_f32_16x16x32_bf16(
          false, a, false, bm, (short)0, hacc, false, false);
    }
    __syncthreads();
  }
  // add bias, deposit h tile (bf16) into LDS using the C/D layout mapping
  {
    const int col = w * 16 + mrow;
    const float bv = bias[col];
    #pragma unroll
    for (int r = 0; r < 8; ++r) {
      const int M = r + half * 8;
      hbf[M * 72 + col] = f2bf(hacc[r] + bv);
    }
  }
  __syncthreads();

  // ------------- stage 2 pass A: softmax denominators per row -------------
  for (int nt = w; nt < 256; nt += 4) {
    const int n0 = nt * 16;
    const int seg = (n0 < 2048) ? 0 : ((n0 < 3072) ? 1 : 2);
    const __bf16* eb = embn + (size_t)(n0 + mrow) * DSZ + half * 16;
    v8f acc = {};
    #pragma unroll
    for (int kb = 0; kb < 2; ++kb) {
      v8bf a0 = *(const v8bf*)&hbf[mrow * 72 + kb * 32 + half * 16];
      v8bf a1 = *(const v8bf*)&hbf[mrow * 72 + kb * 32 + half * 16 + 8];
      v16bf a  = cat8(a0, a1);
      v16bf bm = *(const v16bf*)(eb + kb * 32);
      acc = __builtin_amdgcn_wmma_f32_16x16x32_bf16(
          false, a, false, bm, (short)0, acc, false, false);
    }
    #pragma unroll
    for (int r = 0; r < 8; ++r) {
      float v = __expf(acc[r]);
      v += __shfl_xor(v, 1, 16);
      v += __shfl_xor(v, 2, 16);
      v += __shfl_xor(v, 4, 16);
      v += __shfl_xor(v, 8, 16);
      if (mrow == 0) atomicAdd(&rowSum[seg * 16 + r + half * 8], v);
    }
  }
  __syncthreads();
  if (tid < 48) {
    const int M = tid & 15;
    wrow[tid] = importance[(size_t)b * SEQ + s0 + M] / rowSum[tid];
  }
  __syncthreads();

  // ------- stage 2 pass B: recompute logits, accumulate densities -------
  for (int nt = w; nt < 256; nt += 4) {
    const int n0 = nt * 16;
    const int seg = (n0 < 2048) ? 0 : ((n0 < 3072) ? 1 : 2);
    const __bf16* eb = embn + (size_t)(n0 + mrow) * DSZ + half * 16;
    v8f acc = {};
    #pragma unroll
    for (int kb = 0; kb < 2; ++kb) {
      v8bf a0 = *(const v8bf*)&hbf[mrow * 72 + kb * 32 + half * 16];
      v8bf a1 = *(const v8bf*)&hbf[mrow * 72 + kb * 32 + half * 16 + 8];
      v16bf a  = cat8(a0, a1);
      v16bf bm = *(const v16bf*)(eb + kb * 32);
      acc = __builtin_amdgcn_wmma_f32_16x16x32_bf16(
          false, a, false, bm, (short)0, acc, false, false);
    }
    float part = 0.f;
    #pragma unroll
    for (int r = 0; r < 8; ++r)
      part += __expf(acc[r]) * wrow[seg * 16 + r + half * 8];
    part += __shfl_xor(part, 16, 32);
    if (half == 0)
      atomicAdd(&accum[(size_t)b * NTOT + n0 + mrow], part);
  }
}

// ---------------------------------------------------------------------------
// Final: top-k sparsify per (batch, segment), write concatenated output.
// blockIdx = b*3 + seg; seg1 (dQK) writes both wQ and wK slices.
// ---------------------------------------------------------------------------
__global__ void __launch_bounds__(256)
topk_kernel(const float* __restrict__ accum, float* __restrict__ out) {
  const int b   = blockIdx.x / 3;
  const int seg = blockIdx.x % 3;
  const int segOff = (seg == 0) ? 0 : ((seg == 1) ? 2048 : 3072);
  const int segLen = (seg == 0) ? 2048 : 1024;
  const int K      = (seg == 0) ? 8 : ((seg == 1) ? 4 : 6);
  const int outOff = (seg == 0) ? 0 : ((seg == 1) ? 2048 : 4096);

  __shared__ float buf[2048];
  __shared__ float rv[256];
  __shared__ int   ri[256];
  __shared__ float topVal[8];
  __shared__ int   topIdx[8];

  const int tid = threadIdx.x;
  const float* src = accum + (size_t)b * NTOT + segOff;
  float* ob = out + (size_t)b * 5120;

  for (int i = tid; i < segLen; i += 256) {
    buf[i] = src[i];
    ob[outOff + i] = 0.f;
    if (seg == 1) ob[3072 + i] = 0.f;
  }
  __syncthreads();

  for (int it = 0; it < K; ++it) {
    float best = -1e30f; int bi = 0;
    for (int i = tid; i < segLen; i += 256) {
      const float v = buf[i];
      if (v > best) { best = v; bi = i; }
    }
    rv[tid] = best; ri[tid] = bi;
    __syncthreads();
    for (int s = 128; s >= 1; s >>= 1) {
      if (tid < s && rv[tid + s] > rv[tid]) { rv[tid] = rv[tid + s]; ri[tid] = ri[tid + s]; }
      __syncthreads();
    }
    if (tid == 0) { topVal[it] = rv[0]; topIdx[it] = ri[0]; buf[ri[0]] = -1e30f; }
    __syncthreads();
  }

  if (tid == 0) {
    float s = 0.f;
    for (int i = 0; i < K; ++i) s += topVal[i];
    const float inv = 1.f / (s + 1e-8f);
    for (int i = 0; i < K; ++i) {
      ob[outOff + topIdx[i]] = topVal[i] * inv;
      if (seg == 1) ob[3072 + topIdx[i]] = topVal[i] * inv;
    }
  }
}

// ---------------------------------------------------------------------------
extern "C" void kernel_launch(void* const* d_in, const int* in_sizes, int n_in,
                              void* d_out, int out_size, void* d_ws, size_t ws_size,
                              hipStream_t stream) {
  const float* x          = (const float*)d_in[0];  // (4,4096,2048)
  const float* importance = (const float*)d_in[1];  // (4,4096)
  const float* W          = (const float*)d_in[2];  // (2048,64)
  const float* bias       = (const float*)d_in[3];  // (64,)
  const float* emb        = (const float*)d_in[4];  // (4096,64)

  char* ws = (char*)d_ws;
  __bf16* Wt   = (__bf16*)ws;                        // 64*2048*2  = 256 KB
  __bf16* embn = (__bf16*)(ws + 262144);             // 4096*64*2  = 512 KB
  float*  accum = (float*)(ws + 262144 + 524288);    // 4*4096*4   =  64 KB

  prep_kernel<<<256, 256, 0, stream>>>(W, emb, Wt, embn, accum);
  main_kernel<<<1024, 128, 0, stream>>>(x, importance, Wt, bias, embn, accum);
  topk_kernel<<<12, 256, 0, stream>>>(accum, (float*)d_out);
}
